// NeuralKYAttention_18107582119947
// MI455X (gfx1250) — matrix operand order repaired and checked by reading, not compiled, not run on hardware
//
#include <hip/hip_runtime.h>
#include <hip/hip_bf16.h>

typedef __attribute__((ext_vector_type(16))) __bf16 v16bf;
typedef __attribute__((ext_vector_type(8)))  float  v8f;
typedef __attribute__((ext_vector_type(4)))  unsigned int u32x4;
typedef __attribute__((ext_vector_type(8)))  unsigned int u32x8;

#define D_MODEL 1024
#define NHEAD   16
#define HDIM    64
#define WIN     7
#define SEQ     4096
#define NBATCH  2
#define ROWS    (NBATCH * SEQ)   // 8192

#define KT      64               // K elements staged per pipeline stage
#define LDSTRD  80               // padded LDS row stride (elements); 160B keeps 16B align

// ---------------- helpers ----------------
__device__ __forceinline__ unsigned short f2bf(float f) {
  unsigned int u = __float_as_uint(f);
  u += 0x7fffu + ((u >> 16) & 1u);      // round-to-nearest-even
  return (unsigned short)(u >> 16);
}

__device__ __forceinline__ float gelu_exact(float z) {
  return 0.5f * z * (1.0f + erff(z * 0.70710678118654752f));
}

// ---------------- fp32 -> bf16 conversion ----------------
__global__ void cvt_f32_bf16(const float* __restrict__ src,
                             unsigned short* __restrict__ dst, int n) {
  int i = blockIdx.x * blockDim.x + threadIdx.x;
  if (i < n) dst[i] = f2bf(src[i]);
}

// ---------------- per-head connectivity MLP + softmax over window ----------------
__global__ void conn_kernel(const float* __restrict__ cw1, const float* __restrict__ cb1,
                            const float* __restrict__ cw2, const float* __restrict__ cb2,
                            const float* __restrict__ cw3, const float* __restrict__ cb3,
                            float* __restrict__ conn) {
  int h = threadIdx.x;
  if (h >= NHEAD) return;
  float cwt[WIN];
  for (int w = 0; w < WIN; ++w) {
    float pos = (float)w * (1.0f / 6.0f);   // linspace(0,1,7)
    float h1[32];
    #pragma unroll
    for (int j = 0; j < 32; ++j)
      h1[j] = gelu_exact(pos * cw1[h * 32 + j] + cb1[h * 32 + j]);
    float h2[32];
    for (int j = 0; j < 32; ++j) {
      float s = cb2[h * 32 + j];
      #pragma unroll
      for (int i = 0; i < 32; ++i) s += h1[i] * cw2[(h * 32 + j) * 32 + i];
      h2[j] = gelu_exact(s);
    }
    float s3 = cb3[h];
    #pragma unroll
    for (int i = 0; i < 32; ++i) s3 += h2[i] * cw3[h * 32 + i];
    cwt[w] = s3;
  }
  float mx = cwt[0];
  for (int w = 1; w < WIN; ++w) mx = fmaxf(mx, cwt[w]);
  float e[WIN], sum = 0.0f;
  for (int w = 0; w < WIN; ++w) { e[w] = expf(cwt[w] - mx); sum += e[w]; }
  float inv = 1.0f / sum;
  for (int w = 0; w < WIN; ++w) conn[h * WIN + w] = e[w] * inv;
}

// ---------------- TDM tile load: global (row-major bf16) -> padded LDS tile ----------------
// 2D descriptor per CDNA5 ISA 08_async_tensor: tile = tile_rows x KT elements,
// tensor_dim0_stride = D_MODEL, LDS padding of 8 DWORDs every 32 DWORDs (row
// stride KT=64 elem -> 80 elem in LDS), data_size=2B, type=2 ("image").
__device__ __forceinline__ void tdm_load_tile(unsigned lds_off,
                                              const unsigned short* gptr,
                                              unsigned tile_rows) {
  unsigned long long ga = (unsigned long long)(uintptr_t)gptr;
  u32x4 g0;
  g0[0] = 1u;                                        // count=1, user descriptor
  g0[1] = lds_off;                                   // LDS byte address
  g0[2] = (unsigned)ga;                              // global_addr[31:0]
  g0[3] = (unsigned)((ga >> 32) & 0x1FFFFFFu)        // global_addr[56:32]
        | (2u << 30);                                // type = 2
  u32x8 g1;
  g1[0] = (1u << 16)                                 // data_size: 2 bytes
        | (1u << 20)                                 // pad_enable
        | (4u << 22)                                 // pad_interval: 32 DWORDs
        | (7u << 25);                                // pad_amount: 8 DWORDs
  g1[1] = ((unsigned)D_MODEL & 0xFFFFu) << 16;       // tensor_dim0[15:0]
  g1[2] = ((unsigned)D_MODEL >> 16)                  // tensor_dim0[31:16]
        | ((unsigned)ROWS << 16);                    // tensor_dim1[15:0]
  g1[3] = ((unsigned)ROWS >> 16)                     // tensor_dim1[31:16]
        | ((unsigned)KT << 16);                      // tile_dim0 = 64
  g1[4] = tile_rows;                                 // tile_dim1 (tile_dim2 = 0)
  g1[5] = (unsigned)D_MODEL;                         // tensor_dim0_stride[31:0]
  g1[6] = 0u;
  g1[7] = 0u;
  asm volatile("tensor_load_to_lds %0, %1" :: "s"(g0), "s"(g1) : "memory");
}

// ---------------- bf16 WMMA GEMM: C[M,N] = A[M,K] * W[N,K]^T (+bias) ----------------
// Block tile 128x64, K-stage 64, double-buffered LDS fed by the Tensor Data Mover.
// 8 waves (4 M x 2 N), wave tile 32x32, 8 WMMAs per stage.
__launch_bounds__(256, 2)
__global__ void wmma_gemm(const unsigned short* __restrict__ Abf,
                          const unsigned short* __restrict__ W0,
                          const unsigned short* __restrict__ W1,
                          const unsigned short* __restrict__ W2,
                          float* __restrict__ C0, float* __restrict__ C1,
                          float* __restrict__ C2,
                          const float* __restrict__ bias) {
  const unsigned short* __restrict__ Wbf =
      (blockIdx.z == 0) ? W0 : (blockIdx.z == 1) ? W1 : W2;
  float* __restrict__ C =
      (blockIdx.z == 0) ? C0 : (blockIdx.z == 1) ? C1 : C2;

  __shared__ __align__(16) unsigned short lA[2][128 * LDSTRD];  // 2 x 20.0 KB
  __shared__ __align__(16) unsigned short lB[2][64 * LDSTRD];   // 2 x 10.0 KB

  const int tid  = threadIdx.x;
  const int lane = tid & 31;
  const int wid  = tid >> 5;       // 0..7
  const int wm   = wid >> 1;       // 0..3 (M direction)
  const int wn   = wid & 1;        // 0..1 (N direction)
  const int m0   = blockIdx.y * 128;
  const int n0   = blockIdx.x * 64;
  const int l16  = lane & 15;
  const int half = lane >> 4;      // lane group 0/1

  const unsigned offA0 = (unsigned)(uintptr_t)(&lA[0][0]);
  const unsigned offA1 = (unsigned)(uintptr_t)(&lA[1][0]);
  const unsigned offB0 = (unsigned)(uintptr_t)(&lB[0][0]);
  const unsigned offB1 = (unsigned)(uintptr_t)(&lB[1][0]);

  v8f acc[2][2] = {};

  // Prologue: DMA first K-stage into buffer 0 (one wave drives the TDM).
  if (wid == 0) {
    tdm_load_tile(offA0, Abf + (size_t)m0 * D_MODEL, 128);
    tdm_load_tile(offB0, Wbf + (size_t)n0 * D_MODEL, 64);
    __builtin_amdgcn_s_wait_tensorcnt(0);
  }
  __syncthreads();

  for (int kt = 0; kt < D_MODEL; kt += KT) {
    const int cur = (kt >> 6) & 1;
    const bool more = (kt + KT) < D_MODEL;
    if (wid == 0 && more) {  // DMA next stage into the other buffer (overlaps WMMAs)
      tdm_load_tile(cur ? offA0 : offA1, Abf + (size_t)m0 * D_MODEL + kt + KT, 128);
      tdm_load_tile(cur ? offB0 : offB1, Wbf + (size_t)n0 * D_MODEL + kt + KT, 64);
    }

    const __bf16* pA = (const __bf16*)&lA[cur][0];
    const __bf16* pB = (const __bf16*)&lB[cur][0];
    v16bf af[2][2], bfr[2][2];
    // A frag: lane row = l16; lanes 0-15 hold K{0-7,16-23}, lanes 16-31 K{8-15,24-31}
    #pragma unroll
    for (int mi = 0; mi < 2; ++mi) {
      int row = wm * 32 + mi * 16 + l16;
      #pragma unroll
      for (int kk = 0; kk < 2; ++kk) {
        int base = row * LDSTRD + kk * 32 + half * 8;
        #pragma unroll
        for (int i = 0; i < 8; ++i) {
          af[mi][kk][i]     = pA[base + i];
          af[mi][kk][8 + i] = pA[base + 16 + i];
        }
      }
    }
    // B frag: lane col = l16; lanes 0-15 hold K{0-7,16-23}, lanes 16-31 K{8-15,24-31} (col-major B)
    #pragma unroll
    for (int ni = 0; ni < 2; ++ni) {
      int col = wn * 32 + ni * 16 + l16;
      #pragma unroll
      for (int kk = 0; kk < 2; ++kk) {
        int base = col * LDSTRD + kk * 32 + half * 8;
        #pragma unroll
        for (int i = 0; i < 8; ++i) {
          bfr[ni][kk][i]     = pB[base + i];
          bfr[ni][kk][8 + i] = pB[base + 16 + i];
        }
      }
    }
    #pragma unroll
    for (int kk = 0; kk < 2; ++kk)
      #pragma unroll
      for (int mi = 0; mi < 2; ++mi)
        #pragma unroll
        for (int ni = 0; ni < 2; ++ni)
          acc[mi][ni] = __builtin_amdgcn_wmma_f32_16x16x32_bf16(
              false, af[mi][kk], false, bfr[ni][kk], (short)0, acc[mi][ni],
              false, false);

    if (wid == 0 && more) __builtin_amdgcn_s_wait_tensorcnt(0);
    __syncthreads();   // publish next buffer; protect current buffer from overwrite
  }

  // Epilogue: C layout — col = lane&15, VGPR v -> row v + 8*(lane>=16)
  #pragma unroll
  for (int mi = 0; mi < 2; ++mi) {
    #pragma unroll
    for (int ni = 0; ni < 2; ++ni) {
      int col   = n0 + wn * 32 + ni * 16 + l16;
      int rbase = m0 + wm * 32 + mi * 16 + half * 8;
      float bv = bias ? bias[col] : 0.0f;
      #pragma unroll
      for (int v = 0; v < 8; ++v)
        C[(size_t)(rbase + v) * D_MODEL + col] = acc[mi][ni][v] + bv;
    }
  }
}

// ---------------- sliding-window attention (memory-bound, VALU) ----------------
__launch_bounds__(256)
__global__ void win_attn_kernel(const float* __restrict__ Q, const float* __restrict__ K,
                                const float* __restrict__ V, const float* __restrict__ conn,
                                unsigned short* __restrict__ outbf) {
  int i = blockIdx.x * blockDim.x + threadIdx.x;   // over B*L*H
  int h = i & (NHEAD - 1);
  int l = (i >> 4) & (SEQ - 1);
  int b = i >> 16;
  size_t row = ((size_t)(b * SEQ + l)) * D_MODEL + h * HDIM;

  float4 q4[16];
  const float4* Qp = (const float4*)(Q + row);
  #pragma unroll
  for (int t = 0; t < 16; ++t) q4[t] = Qp[t];

  // scores; reference zero-pads K -> out-of-range window slot contributes score 0
  float s[WIN];
  #pragma unroll
  for (int w = 0; w < WIN; ++w) {
    int kl = l - 3 + w;
    float d = 0.0f;
    if (kl >= 0 && kl < SEQ) {
      const float4* Kp = (const float4*)(K + ((size_t)(b * SEQ + kl)) * D_MODEL + h * HDIM);
      #pragma unroll
      for (int t = 0; t < 16; ++t) {
        float4 kv = Kp[t];
        d += q4[t].x * kv.x + q4[t].y * kv.y + q4[t].z * kv.z + q4[t].w * kv.w;
      }
      d *= 0.125f;   // 1/sqrt(64)
    }
    s[w] = d;
  }
  float mx = s[0];
  #pragma unroll
  for (int w = 1; w < WIN; ++w) mx = fmaxf(mx, s[w]);
  float e[WIN], esum = 0.0f;
  #pragma unroll
  for (int w = 0; w < WIN; ++w) { e[w] = expf(s[w] - mx); esum += e[w]; }
  float fw[WIN], fsum = 0.0f;
  #pragma unroll
  for (int w = 0; w < WIN; ++w) { fw[w] = (e[w] / esum) * conn[h * WIN + w]; fsum += fw[w]; }
  float inv = 1.0f / (fsum + 1e-9f);

  float4 o4[16];
  #pragma unroll
  for (int t = 0; t < 16; ++t) o4[t] = make_float4(0.f, 0.f, 0.f, 0.f);
  #pragma unroll
  for (int w = 0; w < WIN; ++w) {
    int kl = l - 3 + w;
    if (kl < 0 || kl >= SEQ) continue;     // zero-padded V contributes nothing
    float f = fw[w] * inv;
    const float4* Vp = (const float4*)(V + ((size_t)(b * SEQ + kl)) * D_MODEL + h * HDIM);
    #pragma unroll
    for (int t = 0; t < 16; ++t) {
      float4 vv = Vp[t];
      o4[t].x += f * vv.x; o4[t].y += f * vv.y; o4[t].z += f * vv.z; o4[t].w += f * vv.w;
    }
  }
  #pragma unroll
  for (int t = 0; t < 16; ++t) {
    outbf[row + 4 * t + 0] = f2bf(o4[t].x);
    outbf[row + 4 * t + 1] = f2bf(o4[t].y);
    outbf[row + 4 * t + 2] = f2bf(o4[t].z);
    outbf[row + 4 * t + 3] = f2bf(o4[t].w);
  }
}

// ---------------- host launcher ----------------
extern "C" void kernel_launch(void* const* d_in, const int* in_sizes, int n_in,
                              void* d_out, int out_size, void* d_ws, size_t ws_size,
                              hipStream_t stream) {
  const float* x   = (const float*)d_in[0];
  const float* wq  = (const float*)d_in[1];
  const float* wk  = (const float*)d_in[2];
  const float* wv  = (const float*)d_in[3];
  const float* wo  = (const float*)d_in[4];
  const float* bo  = (const float*)d_in[5];
  const float* cw1 = (const float*)d_in[6];
  const float* cb1 = (const float*)d_in[7];
  const float* cw2 = (const float*)d_in[8];
  const float* cb2 = (const float*)d_in[9];
  const float* cw3 = (const float*)d_in[10];
  const float* cb3 = (const float*)d_in[11];
  float* out = (float*)d_out;

  char* ws = (char*)d_ws;
  size_t off = 0;
  unsigned short* xb   = (unsigned short*)(ws + off); off += (size_t)ROWS * D_MODEL * 2;
  unsigned short* wqb  = (unsigned short*)(ws + off); off += (size_t)D_MODEL * D_MODEL * 2;
  unsigned short* wkb  = (unsigned short*)(ws + off); off += (size_t)D_MODEL * D_MODEL * 2;
  unsigned short* wvb  = (unsigned short*)(ws + off); off += (size_t)D_MODEL * D_MODEL * 2;
  unsigned short* wob  = (unsigned short*)(ws + off); off += (size_t)D_MODEL * D_MODEL * 2;
  float*          Qf   = (float*)(ws + off);          off += (size_t)ROWS * D_MODEL * 4;
  float*          Kf   = (float*)(ws + off);          off += (size_t)ROWS * D_MODEL * 4;
  float*          Vf   = (float*)(ws + off);          off += (size_t)ROWS * D_MODEL * 4;
  unsigned short* aob  = (unsigned short*)(ws + off); off += (size_t)ROWS * D_MODEL * 2;
  float*          conw = (float*)(ws + off);          off += (size_t)NHEAD * WIN * 4;

  // 1) precision staging: fp32 -> bf16 for GEMM operands
  int nx = ROWS * D_MODEL;
  cvt_f32_bf16<<<(nx + 255) / 256, 256, 0, stream>>>(x, xb, nx);
  int nw = D_MODEL * D_MODEL;
  cvt_f32_bf16<<<(nw + 255) / 256, 256, 0, stream>>>(wq, wqb, nw);
  cvt_f32_bf16<<<(nw + 255) / 256, 256, 0, stream>>>(wk, wkb, nw);
  cvt_f32_bf16<<<(nw + 255) / 256, 256, 0, stream>>>(wv, wvb, nw);
  cvt_f32_bf16<<<(nw + 255) / 256, 256, 0, stream>>>(wo, wob, nw);

  // 2) tiny per-head connectivity MLP
  conn_kernel<<<1, 16, 0, stream>>>(cw1, cb1, cw2, cb2, cw3, cb3, conw);

  // 3) fused Q/K/V projections: grid.z selects weight/output
  dim3 gQKV(D_MODEL / 64, ROWS / 128, 3);
  wmma_gemm<<<gQKV, 256, 0, stream>>>(xb, wqb, wkb, wvb, Qf, Kf, Vf, nullptr);

  // 4) sliding-window attention -> bf16 activations
  int na = NBATCH * SEQ * NHEAD;
  win_attn_kernel<<<na / 256, 256, 0, stream>>>(Qf, Kf, Vf, conw, aob);

  // 5) output projection + bias -> d_out (fp32)
  dim3 gO(D_MODEL / 64, ROWS / 128, 1);
  wmma_gemm<<<gO, 256, 0, stream>>>(aob, wob, wob, wob, out, out, out, bo);

  (void)in_sizes; (void)n_in; (void)out_size; (void)ws_size;
}
